// MultiHeadAttention_19593640804983
// MI455X (gfx1250) — matrix operand order repaired and checked by reading, not compiled, not run on hardware
//
#include <hip/hip_runtime.h>
#include <hip/hip_bf16.h>

// Problem constants (from reference)
#define Bq 4
#define Tq 2048
#define Cq 2048
#define Hq 16
#define Dq 128
#define BT (Bq * Tq)

typedef __bf16 bf16_t;
typedef __attribute__((ext_vector_type(4)))  __bf16 v4bf;
typedef __attribute__((ext_vector_type(8)))  __bf16 v8bf;
typedef __attribute__((ext_vector_type(16))) __bf16 v16bf;
typedef __attribute__((ext_vector_type(8)))  float  v8f;

union Frag16 { v16bf v; v8bf h[2]; };

static __device__ __forceinline__ v8f vzero8() {
  v8f z;
#pragma unroll
  for (int i = 0; i < 8; ++i) z[i] = 0.0f;
  return z;
}

static __device__ __forceinline__ v8f wmma_bf16(v16bf a, v16bf b, v8f c) {
  // D = A(16x32 bf16) * B(32x16 bf16) + C(16x16 f32)
  return __builtin_amdgcn_wmma_f32_16x16x32_bf16(false, a, false, b,
                                                 (short)0, c, false, false);
}

// Async global -> LDS copy of 16 bytes (ASYNCcnt-tracked; bypasses VGPRs).
// VDST operand carries the LDS byte offset: generic pointers to LDS hold the
// LDS offset in addr[31:0] (ISA aperture mapping), so truncation is exact.
static __device__ __forceinline__ void async_ld128(const bf16_t* gsrc,
                                                   bf16_t* ldst) {
  unsigned loff = (unsigned)(unsigned long long)(void*)ldst;
  asm volatile("global_load_async_to_lds_b128 %0, %1, off"
               :: "v"(loff), "v"(gsrc)
               : "memory");
}

static __device__ __forceinline__ void wait_async0() {
  asm volatile("s_wait_asynccnt 0x0" ::: "memory");
}

// ---------------------------------------------------------------------------
// Stage 0: fp32 -> bf16 conversion (vectorized x4)
// ---------------------------------------------------------------------------
__global__ void k_cvt_bf16(const float* __restrict__ in,
                           bf16_t* __restrict__ out, int n4) {
  int i = blockIdx.x * blockDim.x + threadIdx.x;
  if (i >= n4) return;
  float4 f = ((const float4*)in)[i];
  v4bf o;
  o[0] = (bf16_t)f.x; o[1] = (bf16_t)f.y;
  o[2] = (bf16_t)f.z; o[3] = (bf16_t)f.w;
  ((v4bf*)out)[i] = o;
}

// ---------------------------------------------------------------------------
// Stage 0b: RoPE tables cos/sin[t][j], j in [0,64), theta = 1e6
// ---------------------------------------------------------------------------
__global__ void k_rope_tab(float* __restrict__ c, float* __restrict__ s) {
  int i = blockIdx.x * blockDim.x + threadIdx.x; // Tq*64 threads
  int t = i >> 6, j = i & 63;
  float inv = powf(1.0e6f, -(float)j * (1.0f / 64.0f));
  float f = (float)t * inv;
  c[i] = cosf(f);
  s[i] = sinf(f);
}

// ---------------------------------------------------------------------------
// bf16 GEMM: out[m,n] = sum_k A[m,k]*Bw[n,k]   (i.e. A @ Bw^T)
// Block tile 128x128, staged BK=64 (two K=32 WMMA sub-steps per stage),
// 256 threads = 8 waves (4x2), wave tile 32x64 -> 16 WMMAs per stage per wave.
// Double-buffered LDS (2 x 32 KB) with async global->LDS staging.
// MODE 0: Q  -> rope + 1/sqrt(D) scale, bf16 out in (B,H,T,D)
// MODE 1: K  -> rope, bf16 out in (B,H,T,D)
// MODE 2: V  -> bf16 out in (B,H,T,D)
// MODE 3: final projection -> fp32 out (BT,C) + bias
// ---------------------------------------------------------------------------
template<int MODE>
__global__ __launch_bounds__(256) void k_gemm(
    const bf16_t* __restrict__ A, const bf16_t* __restrict__ Bw,
    bf16_t* __restrict__ outB, float* __restrict__ outF,
    const float* __restrict__ bias,
    const float* __restrict__ rcos, const float* __restrict__ rsin,
    int Kdim)
{
  __shared__ __align__(16) bf16_t sA[2][128 * 64];
  __shared__ __align__(16) bf16_t sB[2][128 * 64];

  const int tid    = threadIdx.x;
  const int mBlk   = blockIdx.y * 128;
  const int nBlk   = blockIdx.x * 128;
  const int wave   = tid >> 5;
  const int lane   = tid & 31;
  const int laneN  = lane & 15;
  const int laneHi = lane >> 4;
  const int wm     = wave >> 1;   // 0..3
  const int wn     = wave & 1;    // 0..1

  v8f acc[2][4];
#pragma unroll
  for (int i = 0; i < 2; ++i)
#pragma unroll
    for (int j = 0; j < 4; ++j) acc[i][j] = vzero8();

  // Staging map: thread t -> row = t/2 (of 128), 32-element half-row (t&1)
  const int ldRow = tid >> 1;
  const int ldCol = (tid & 1) * 32;
  const bf16_t* gA = A  + (size_t)(mBlk + ldRow) * Kdim + ldCol;
  const bf16_t* gB = Bw + (size_t)(nBlk + ldRow) * Kdim + ldCol;
  bf16_t* lA[2] = { &sA[0][ldRow * 64 + ldCol], &sA[1][ldRow * 64 + ldCol] };
  bf16_t* lB[2] = { &sB[0][ldRow * 64 + ldCol], &sB[1][ldRow * 64 + ldCol] };

  // Prologue: stage K-tile 0 into buffer 0 (4 async b128 per matrix)
#pragma unroll
  for (int u = 0; u < 4; ++u) {
    async_ld128(gA + u * 8, lA[0] + u * 8);
    async_ld128(gB + u * 8, lB[0] + u * 8);
  }

  int buf = 0;
  for (int k0 = 0; k0 < Kdim; k0 += 64, buf ^= 1) {
    wait_async0();       // staged tile (issued last iteration) resident in LDS
    __syncthreads();     // visible to all waves; prev tile's frag reads done

    if (k0 + 64 < Kdim) {
      const int kn = k0 + 64;
#pragma unroll
      for (int u = 0; u < 4; ++u) {
        async_ld128(gA + kn + u * 8, lA[buf ^ 1] + u * 8);
        async_ld128(gB + kn + u * 8, lB[buf ^ 1] + u * 8);
      }
    }

    const bf16_t* cA = &sA[buf][0];
    const bf16_t* cB = &sB[buf][0];
#pragma unroll
    for (int kk = 0; kk < 64; kk += 32) {
      Frag16 af[2], bfr[4];
#pragma unroll
      for (int i = 0; i < 2; ++i) {
        int ar = wm * 32 + i * 16 + laneN;
        int kb = kk + laneHi * 16; // lanes 0-15: K 0..15; lanes 16-31: K 16..31
        af[i].h[0] = *(const v8bf*)&cA[ar * 64 + kb];
        af[i].h[1] = *(const v8bf*)&cA[ar * 64 + kb + 8];
      }
#pragma unroll
      for (int j = 0; j < 4; ++j) {
        int br = wn * 64 + j * 16 + laneN;
        int kb = kk + laneHi * 16; // lanes 0-15: K 0..15; lanes 16-31: K 16..31
        bfr[j].h[0] = *(const v8bf*)&cB[br * 64 + kb];
        bfr[j].h[1] = *(const v8bf*)&cB[br * 64 + kb + 8];
      }
#pragma unroll
      for (int i = 0; i < 2; ++i)
#pragma unroll
        for (int j = 0; j < 4; ++j)
          acc[i][j] = wmma_bf16(af[i].v, bfr[j].v, acc[i][j]);
    }
  }

  // Epilogue. C/D layout: VGPR e, lanes 0-15 -> M=e; lanes 16-31 -> M=e+8; N=lane&15
#pragma unroll
  for (int i = 0; i < 2; ++i)
#pragma unroll
    for (int j = 0; j < 4; ++j)
#pragma unroll
      for (int e = 0; e < 8; ++e) {
        int r = mBlk + wm * 32 + i * 16 + e + laneHi * 8;
        int c = nBlk + wn * 64 + j * 16 + laneN;
        float val = acc[i][j][e];
        if (MODE == 3) {
          outF[(size_t)r * Cq + c] = val + bias[c];
        } else {
          int t  = r & (Tq - 1);
          int bb = r >> 11;        // / Tq
          int hh = c >> 7;         // / Dq
          int d  = c & 127;
          float res;
          if (MODE <= 1) {
            // RoPE: pair partner lives in adjacent lane (c ^ 1)
            float partner = __shfl_xor(val, 1);
            int j2 = d >> 1;
            float cv = rcos[t * 64 + j2];
            float sv = rsin[t * 64 + j2];
            res = (c & 1) ? (val * cv + partner * sv)
                          : (val * cv - partner * sv);
            if (MODE == 0) res *= 0.08838834764831845f; // 1/sqrt(128)
          } else {
            res = val;
          }
          outB[(((size_t)(bb * Hq + hh)) * Tq + t) * Dq + d] = (bf16_t)res;
        }
      }
}

// ---------------------------------------------------------------------------
// Flash attention, causal. One block per (b*H+h, 128-query tile).
// 8 waves x 16 query rows. Key tiles of 32. Q prescaled by 1/sqrt(D).
// K tile staged via async global->LDS; V staged with in-register transpose.
// ---------------------------------------------------------------------------
#define MASKVAL (-3.0e38f)

__global__ __launch_bounds__(256) void k_flash_attn(
    const bf16_t* __restrict__ Q, const bf16_t* __restrict__ Kb,
    const bf16_t* __restrict__ Vb, bf16_t* __restrict__ O)
{
  __shared__ __align__(16) bf16_t sK[32 * 128];   // key rows x D
  __shared__ __align__(16) bf16_t sVt[128 * 32];  // D x key rows (transposed)
  __shared__ __align__(16) bf16_t sP[8][16 * 32]; // per-wave P scratch

  const int tid    = threadIdx.x;
  const int wave   = tid >> 5;
  const int lane   = tid & 31;
  const int laneN  = lane & 15;
  const int laneHi = lane >> 4;

  const int bh    = blockIdx.y;      // b*H + h
  const int b     = bh >> 4;
  const int h     = bh & 15;
  const int qBase = blockIdx.x * 128;
  const int qRow0 = qBase + wave * 16;

  const bf16_t* Qp = Q  + (size_t)bh * Tq * Dq;
  const bf16_t* Kp = Kb + (size_t)bh * Tq * Dq;
  const bf16_t* Vp = Vb + (size_t)bh * Tq * Dq;

  // Q fragments for this wave's 16 rows, 4 chunks of K=32 over D=128
  Frag16 qf[4];
  {
    const bf16_t* qr = Qp + (size_t)(qRow0 + laneN) * Dq;
#pragma unroll
    for (int kc = 0; kc < 4; ++kc) {
      int kb = kc * 32 + laneHi * 16;
      qf[kc].h[0] = *(const v8bf*)(qr + kb);
      qf[kc].h[1] = *(const v8bf*)(qr + kb + 8);
    }
  }

  v8f o[8];
#pragma unroll
  for (int j = 0; j < 8; ++j) o[j] = vzero8();
  float mRun[8], lRun[8];
#pragma unroll
  for (int e = 0; e < 8; ++e) { mRun[e] = MASKVAL; lRun[e] = 0.0f; }

  const int ldRowKV = tid >> 3;        // 0..31 key row in tile
  const int ldSeg   = (tid & 7) * 16;  // 16-elem column segment

  const int ktEnd = (qBase + 128) >> 5;
  for (int kt = 0; kt < ktEnd; ++kt) {
    const int k0 = kt << 5;
    __syncthreads();
    {
      // K tile: async global -> LDS (row-major)
      const bf16_t* gk = Kp + (size_t)(k0 + ldRowKV) * Dq + ldSeg;
      bf16_t* lk = &sK[ldRowKV * 128 + ldSeg];
      async_ld128(gk,     lk);
      async_ld128(gk + 8, lk + 8);
      // V tile: load to regs, store transposed into LDS
      const bf16_t* gv = Vp + (size_t)(k0 + ldRowKV) * Dq + ldSeg;
      v8bf a0 = *(const v8bf*)gv;
      v8bf a1 = *(const v8bf*)(gv + 8);
#pragma unroll
      for (int u = 0; u < 8; ++u) {
        sVt[(ldSeg + u) * 32 + ldRowKV]     = a0[u];
        sVt[(ldSeg + 8 + u) * 32 + ldRowKV] = a1[u];
      }
    }
    wait_async0();
    __syncthreads();

    // S = Q K^T : 16 x 32 (two 16x16 tiles)
    v8f sL = vzero8(), sR = vzero8();
#pragma unroll
    for (int kc = 0; kc < 4; ++kc) {
      Frag16 bL, bR;
      int kb = kc * 32 + laneHi * 16;
      const bf16_t* kr  = &sK[laneN * 128 + kb];
      const bf16_t* kr2 = &sK[(16 + laneN) * 128 + kb];
      bL.h[0] = *(const v8bf*)kr;   bL.h[1] = *(const v8bf*)(kr + 8);
      bR.h[0] = *(const v8bf*)kr2;  bR.h[1] = *(const v8bf*)(kr2 + 8);
      sL = wmma_bf16(qf[kc].v, bL.v, sL);
      sR = wmma_bf16(qf[kc].v, bR.v, sR);
    }

    // Online softmax update
    float pL[8], pR[8], alpha[8];
#pragma unroll
    for (int e = 0; e < 8; ++e) {
      int r  = qRow0 + e + laneHi * 8;       // global query index
      int kL = k0 + laneN;
      int kR = k0 + 16 + laneN;
      float xL = (kL <= r) ? sL[e] : MASKVAL;
      float xR = (kR <= r) ? sR[e] : MASKVAL;
      float mt = fmaxf(xL, xR);
#pragma unroll
      for (int off = 1; off < 16; off <<= 1)
        mt = fmaxf(mt, __shfl_xor(mt, off));
      float mNew = fmaxf(mRun[e], mt);
      float a  = __expf(mRun[e] - mNew);
      float eL = __expf(xL - mNew);
      float eR = __expf(xR - mNew);
      float rs = eL + eR;
#pragma unroll
      for (int off = 1; off < 16; off <<= 1) rs += __shfl_xor(rs, off);
      lRun[e]  = lRun[e] * a + rs;
      mRun[e]  = mNew;
      alpha[e] = a;
      pL[e] = eL; pR[e] = eR;
    }
#pragma unroll
    for (int j = 0; j < 8; ++j)
#pragma unroll
      for (int e = 0; e < 8; ++e) o[j][e] = o[j][e] * alpha[e];

    // P (C-layout) -> LDS -> A-fragment layout
    bf16_t* pb = &sP[wave][0];
#pragma unroll
    for (int e = 0; e < 8; ++e) {
      int row = e + laneHi * 8;
      pb[row * 32 + laneN]      = (bf16_t)pL[e];
      pb[row * 32 + 16 + laneN] = (bf16_t)pR[e];
    }
    Frag16 pf;
    {
      int kb = laneHi * 16;
      pf.h[0] = *(const v8bf*)&pb[laneN * 32 + kb];
      pf.h[1] = *(const v8bf*)&pb[laneN * 32 + kb + 8];
    }

    // O += P * V   (V transposed in LDS: B-frag reads contiguous key index)
#pragma unroll
    for (int j = 0; j < 8; ++j) {
      Frag16 vf;
      int drow = j * 16 + laneN;
      int kb = laneHi * 16;
      vf.h[0] = *(const v8bf*)&sVt[drow * 32 + kb];
      vf.h[1] = *(const v8bf*)&sVt[drow * 32 + kb + 8];
      o[j] = wmma_bf16(pf.v, vf.v, o[j]);
    }
  }

  // Normalize and store bf16 into (B,T,C) with c = h*D + d
#pragma unroll
  for (int e = 0; e < 8; ++e) {
    float inv = 1.0f / lRun[e];
    int r = qRow0 + e + laneHi * 8;
    size_t base = ((size_t)b * Tq + r) * Cq + h * Dq;
#pragma unroll
    for (int j = 0; j < 8; ++j)
      O[base + j * 16 + laneN] = (bf16_t)(o[j][e] * inv);
  }
}

// ---------------------------------------------------------------------------
// Host-side launcher
// ---------------------------------------------------------------------------
extern "C" void kernel_launch(void* const* d_in, const int* in_sizes, int n_in,
                              void* d_out, int out_size, void* d_ws, size_t ws_size,
                              hipStream_t stream) {
  (void)in_sizes; (void)n_in; (void)out_size; (void)ws_size;
  const float* x  = (const float*)d_in[0];
  const float* Wq = (const float*)d_in[1];
  const float* Wk = (const float*)d_in[2];
  const float* Wv = (const float*)d_in[3];
  const float* Wo = (const float*)d_in[4];
  const float* bo = (const float*)d_in[5];
  float* out = (float*)d_out;

  const size_t MB = 1024ull * 1024ull;
  char* ws = (char*)d_ws;
  bf16_t* Xb  = (bf16_t*)(ws);              // 32 MB  (BT*C bf16)
  bf16_t* Wqb = (bf16_t*)(ws + 32  * MB);   //  8 MB
  bf16_t* Wkb = (bf16_t*)(ws + 40  * MB);
  bf16_t* Wvb = (bf16_t*)(ws + 48  * MB);
  bf16_t* Wob = (bf16_t*)(ws + 56  * MB);
  bf16_t* Qb  = (bf16_t*)(ws + 64  * MB);   // 32 MB (B,H,T,D)
  bf16_t* Kb  = (bf16_t*)(ws + 96  * MB);
  bf16_t* Vb  = (bf16_t*)(ws + 128 * MB);
  bf16_t* Ob  = (bf16_t*)(ws + 160 * MB);   // 32 MB (BT,C)
  float*  rC  = (float*) (ws + 192 * MB);   // T*64 f32
  float*  rS  = rC + Tq * 64;

  // Stage 0: conversions + rope tables
  k_cvt_bf16<<<(BT * Cq / 4 + 255) / 256, 256, 0, stream>>>(x, Xb, BT * Cq / 4);
  k_cvt_bf16<<<(Cq * Cq / 4 + 255) / 256, 256, 0, stream>>>(Wq, Wqb, Cq * Cq / 4);
  k_cvt_bf16<<<(Cq * Cq / 4 + 255) / 256, 256, 0, stream>>>(Wk, Wkb, Cq * Cq / 4);
  k_cvt_bf16<<<(Cq * Cq / 4 + 255) / 256, 256, 0, stream>>>(Wv, Wvb, Cq * Cq / 4);
  k_cvt_bf16<<<(Cq * Cq / 4 + 255) / 256, 256, 0, stream>>>(Wo, Wob, Cq * Cq / 4);
  k_rope_tab<<<(Tq * 64) / 256, 256, 0, stream>>>(rC, rS);

  // Stage 1: QKV projections (fused RoPE / scale)
  dim3 gGemm(Cq / 128, BT / 128);
  k_gemm<0><<<gGemm, 256, 0, stream>>>(Xb, Wqb, Qb, nullptr, nullptr, rC, rS, Cq);
  k_gemm<1><<<gGemm, 256, 0, stream>>>(Xb, Wkb, Kb, nullptr, nullptr, rC, rS, Cq);
  k_gemm<2><<<gGemm, 256, 0, stream>>>(Xb, Wvb, Vb, nullptr, nullptr, rC, rS, Cq);

  // Stage 2: causal flash attention
  dim3 gAttn(Tq / 128, Bq * Hq);
  k_flash_attn<<<gAttn, 256, 0, stream>>>(Qb, Kb, Vb, Ob);

  // Stage 3: output projection + bias (fp32 out)
  k_gemm<3><<<gGemm, 256, 0, stream>>>(Ob, Wob, nullptr, out, bo, nullptr, nullptr, Cq);
}
